// Mamba3LBlock_2353642078643
// MI455X (gfx1250) — matrix operand order repaired and checked by reading, not compiled, not run on hardware
//
#include <hip/hip_runtime.h>

// ---------------- model dims ----------------
#define D_MODEL 1024
#define D_INNER 2048
#define NHEADS  32
#define D_STATE 128
#define RANK    4
#define SUBDIM  16
#define NCOLS   5152   // 2*D_INNER + 2*RANK*D_STATE + NHEADS
#define OFF_Z   0
#define OFF_X   2048
#define OFF_B   4096
#define OFF_C   4608
#define OFF_DT  5120
#define BATCH   4
#define SEQLEN  2048
#define NROWS   (BATCH * SEQLEN)   // 8192

typedef __attribute__((ext_vector_type(16))) __bf16 v16bf;
typedef __attribute__((ext_vector_type(8)))  float  v8f;
typedef __attribute__((ext_vector_type(2)))  float  v2f;
typedef __attribute__((ext_vector_type(4)))  int    v4i;

// async-to-LDS builtin takes <4 x i32> pointers: global (AS1) src, LDS (AS3) dst
typedef __attribute__((address_space(1))) v4i gv4i;
typedef __attribute__((address_space(3))) v4i sv4i;

// Detect gfx1250 async global->LDS builtins (compile-only probe via #warning)
#if defined(__HIP_DEVICE_COMPILE__)
#if __has_builtin(__builtin_amdgcn_global_load_async_to_lds_b128)
#define HAVE_ASYNC_LDS 1
#else
#define HAVE_ASYNC_LDS 0
#warning "gfx1250 async-to-LDS builtin NOT available; using synchronous staging fallback"
#endif
#else
#define HAVE_ASYNC_LDS 0
#endif

__device__ __forceinline__ void wait_async0() {
#if HAVE_ASYNC_LDS
#if __has_builtin(__builtin_amdgcn_s_wait_asynccnt)
  __builtin_amdgcn_s_wait_asynccnt(0);
#else
  asm volatile("s_wait_asynccnt 0" ::: "memory");
#endif
#endif
}

__device__ __forceinline__ unsigned short f2bf(float f) {
  unsigned int u = __float_as_uint(f);
  u += 0x7FFFu + ((u >> 16) & 1u);          // round-to-nearest-even
  return (unsigned short)(u >> 16);
}

// ---------------- elementwise kernels ----------------
__global__ __launch_bounds__(256)
void add_cvt_kernel(const float* __restrict__ a, const float* __restrict__ b,
                    float* __restrict__ hs, unsigned short* __restrict__ hsb, int n) {
  int i = blockIdx.x * blockDim.x + threadIdx.x;
  if (i < n) {
    float v = a[i] + b[i];
    hs[i]  = v;
    hsb[i] = f2bf(v);
  }
}

__global__ __launch_bounds__(256)
void cvt_kernel(const float* __restrict__ a, unsigned short* __restrict__ o, int n) {
  int i = blockIdx.x * blockDim.x + threadIdx.x;
  if (i < n) o[i] = f2bf(a[i]);
}

// ---------------- bf16 WMMA GEMM:  C[M,N] = A[M,K] * B[K,N]  (f32 accum) ----------------
// Block tile 128x128, 8 waves (wave32), wave tile 64x32 = 4x2 wmma 16x16 tiles, K-step 32.
// Double-buffered LDS; A staged via async global->LDS (ASYNCcnt), B staged transposed.
union FragBF { uint4 q[2]; v16bf v; };

__global__ __launch_bounds__(256)
void gemm_bf16_wmma(const unsigned short* __restrict__ A,
                    const unsigned short* __restrict__ B,
                    float* __restrict__ C, int M, int N, int K) {
  __shared__ __align__(16) unsigned short sA[2][128 * 32];   // A tiles [128 m][32 k]
  __shared__ __align__(16) unsigned short sBt[2][128 * 32];  // B tiles transposed [128 n][32 k]

  const int tid  = threadIdx.x;
  const int lane = tid & 31;
  const int wave = tid >> 5;
  const int wm = wave >> 2;              // 0..1
  const int wn = wave & 3;               // 0..3
  const int bm = blockIdx.y * 128;
  const int bn = blockIdx.x * 128;
  const int lm = lane & 15;
  const int kbA = (lane < 16) ? 0 : 8;   // 16-bit A layout half-wave K-base
  const int kbB = (lane < 16) ? 0 : 8;   // 16-bit B layout half-wave K-base

  const int ar = tid >> 1;               // A: 0..127 row
  const int ac = (tid & 1) * 16;         // A: 0 or 16 (16 elements = 32B per thread)
  const int kp = tid >> 4;               // B: k-row pair 0..15
  const int nb = (tid & 15) * 8;         // B: n-base 0..120

  v8f acc[4][2];
#pragma unroll
  for (int i = 0; i < 4; ++i)
#pragma unroll
    for (int j = 0; j < 2; ++j)
#pragma unroll
      for (int v = 0; v < 8; ++v) acc[i][j][v] = 0.f;

  auto stageA = [&](int buf, int kk) {
    const unsigned short* g = A + (size_t)(bm + ar) * K + kk + ac;
    unsigned short* l = &sA[buf][ar * 32 + ac];
#if HAVE_ASYNC_LDS
    __builtin_amdgcn_global_load_async_to_lds_b128((gv4i*)g, (sv4i*)l, 0, 0);
    __builtin_amdgcn_global_load_async_to_lds_b128((gv4i*)(g + 8), (sv4i*)(l + 8), 0, 0);
#else
    *(uint4*)l       = *(const uint4*)g;
    *(uint4*)(l + 8) = *(const uint4*)(g + 8);
#endif
  };
  auto stageB = [&](int buf, int kk) {
    // load two consecutive k-rows, store packed (k,k+1) pairs -> transposed LDS layout
    const unsigned short* s0 = B + (size_t)(kk + 2 * kp) * N + bn + nb;
    const unsigned short* s1 = s0 + N;
    __align__(16) unsigned short r0[8], r1[8];
    if (bn + nb + 8 <= N) {
      *(uint4*)r0 = *(const uint4*)s0;
      *(uint4*)r1 = *(const uint4*)s1;
    } else {
#pragma unroll
      for (int j = 0; j < 8; ++j) {
        r0[j] = (bn + nb + j < N) ? s0[j] : (unsigned short)0;
        r1[j] = (bn + nb + j < N) ? s1[j] : (unsigned short)0;
      }
    }
#pragma unroll
    for (int j = 0; j < 8; ++j) {
      unsigned int pk = (unsigned int)r0[j] | ((unsigned int)r1[j] << 16);
      *(unsigned int*)&sBt[buf][(nb + j) * 32 + 2 * kp] = pk;
    }
  };

  const int nk = K >> 5;
  stageA(0, 0);
  stageB(0, 0);

  for (int ik = 0; ik < nk; ++ik) {
    const int buf = ik & 1;
    wait_async0();          // my async writes into sA[buf] done
    __syncthreads();        // everyone's staging of buf visible

    if (ik + 1 < nk) {      // issue next tile into the other buffer while computing
      stageA(buf ^ 1, (ik + 1) * 32);
      stageB(buf ^ 1, (ik + 1) * 32);
      if (ik + 2 < nk)      // L2 warm hint two tiles ahead (global_prefetch_b8)
        __builtin_prefetch(A + (size_t)(bm + ar) * K + (ik + 2) * 32 + ac, 0, 1);
    }

    const unsigned short* a0 = &sA[buf][0];
    const unsigned short* b0 = &sBt[buf][0];
    FragBF af[4], bfr[2];
#pragma unroll
    for (int i = 0; i < 4; ++i) {
      int m = wm * 64 + i * 16 + lm;
      af[i].q[0] = *(const uint4*)&a0[m * 32 + kbA];        // K = kb..kb+7
      af[i].q[1] = *(const uint4*)&a0[m * 32 + kbA + 16];   // K = kb+16..kb+23
    }
#pragma unroll
    for (int j = 0; j < 2; ++j) {
      int n = wn * 32 + j * 16 + lm;
      bfr[j].q[0] = *(const uint4*)&b0[n * 32 + kbB];       // K = kbB..kbB+7
      bfr[j].q[1] = *(const uint4*)&b0[n * 32 + kbB + 16];  // K = kbB+16..kbB+23
    }
#pragma unroll
    for (int i = 0; i < 4; ++i)
#pragma unroll
      for (int j = 0; j < 2; ++j)
        acc[i][j] = __builtin_amdgcn_wmma_f32_16x16x32_bf16(
            false, af[i].v, false, bfr[j].v, (short)0, acc[i][j], false, false);
    // no trailing barrier needed: next iteration writes the other buffer
  }

  const int rowAdj = (lane < 16) ? 0 : 8;   // D layout: VGPR v -> M = v (+8 upper half)
#pragma unroll
  for (int i = 0; i < 4; ++i)
#pragma unroll
    for (int j = 0; j < 2; ++j) {
      int ncol = bn + wn * 32 + j * 16 + lm;
      if (ncol < N) {
        int mbase = bm + wm * 64 + i * 16 + rowAdj;
#pragma unroll
        for (int v = 0; v < 8; ++v)
          C[(size_t)(mbase + v) * N + ncol] = acc[i][j][v];
      }
    }
}

// ---------------- selective scan: one wave per (b,h), state h[128,16] in 8 wmma tiles ----
// h_t = a*h + wmma_f32_16x16x4(dt*B^T_tile, x)   ;   y = C^T h via VALU reduction
// fused epilogue: ybar = (y + D*x) * silu(z)  -> bf16
__global__ __launch_bounds__(1024)
void scan_kernel(const float* __restrict__ zx,
                 const float* __restrict__ A_log,
                 const float* __restrict__ dt_bias,
                 const float* __restrict__ D_skip,
                 unsigned short* __restrict__ ybar) {
  __shared__ __align__(16) float  sBC[1024];    // B row (512) | C row (512), shared by heads
  __shared__ __align__(16) float4 sPart[1024];  // per-lane partial y, half-wave exchange

  const int b    = blockIdx.x;
  const int tid  = threadIdx.x;
  const int lane = tid & 31;
  const int h    = tid >> 5;                 // wave == head
  const int p    = lane & 15;                // column within 16-wide tile
  const int ka   = (lane < 16) ? 0 : 2;      // f32 wmma A/B layout half-wave K-base
  const int hOff = (lane < 16) ? 0 : 8;      // D layout row adj

  const float negA = -__expf(A_log[h]);
  const float bias = dt_bias[h];
  const float dsk  = D_skip[h];

  v8f hst[8];                                // state: 8 tiles of 16(n) x 16(p), fp32
#pragma unroll
  for (int t = 0; t < 8; ++t)
#pragma unroll
    for (int v = 0; v < 8; ++v) hst[t][v] = 0.f;

  for (int step = 0; step < SEQLEN; ++step) {
    const size_t rb = ((size_t)b * SEQLEN + step) * NCOLS;
    sBC[tid] = zx[rb + OFF_B + tid];         // cooperative: B[4,128] then C[4,128]
    __syncthreads();

    // x fragment = B-matrix operand of 16x16x4 f32 wmma: lane holds x[ka,p], x[ka+1,p]
    const float* xr = zx + rb + OFF_X + (size_t)h * 64;
    v2f xf;
    xf[0] = xr[ka * 16 + p];
    xf[1] = xr[(ka + 1) * 16 + p];

    float dtr = zx[rb + OFF_DT + h] + bias;
    float dtv = (dtr > 20.f) ? dtr : log1pf(__expf(dtr));   // softplus
    float a   = __expf(negA * dtv);                         // per-head decay

    const float* sB = sBC;
    const float* sC = sBC + 512;

#pragma unroll
    for (int t = 0; t < 8; ++t) {
      // A fragment: A[m=n_local, k=r] = dt * B[r, t*16+m]
      v2f afr;
      afr[0] = dtv * sB[ka * 128 + t * 16 + p];
      afr[1] = dtv * sB[(ka + 1) * 128 + t * 16 + p];
#pragma unroll
      for (int v = 0; v < 8; ++v) hst[t][v] *= a;           // decay state
      hst[t] = __builtin_amdgcn_wmma_f32_16x16x4_f32(
          false, afr, false, xf, (short)0, hst[t], false, false);
    }

    // y[r,p] = sum_n C[r,n]*h[n,p]; this lane owns n = t*16 + v + hOff, col p
    float part[4] = {0.f, 0.f, 0.f, 0.f};
#pragma unroll
    for (int t = 0; t < 8; ++t)
#pragma unroll
      for (int r = 0; r < 4; ++r) {
        const float* cp = sC + r * 128 + t * 16 + hOff;     // 16B-aligned broadcast
        float4 c0 = *(const float4*)cp;
        float4 c1 = *(const float4*)(cp + 4);
        part[r] += c0.x * hst[t][0] + c0.y * hst[t][1] + c0.z * hst[t][2] + c0.w * hst[t][3]
                 + c1.x * hst[t][4] + c1.y * hst[t][5] + c1.z * hst[t][6] + c1.w * hst[t][7];
      }

    // combine lane p with lane p+16 (same wave -> LDS ops in order, no barrier needed)
    float4 mine = {part[0], part[1], part[2], part[3]};
    sPart[tid] = mine;
    float4 oth = sPart[tid ^ 16];
    float t0 = (lane < 16) ? (mine.x + oth.x) : (mine.z + oth.z);  // r = ka
    float t1 = (lane < 16) ? (mine.y + oth.y) : (mine.w + oth.w);  // r = ka+1

    // fused gate: (y + D*x) * silu(z) -> bf16 ; lane writes its two rank streams
#pragma unroll
    for (int u = 0; u < 2; ++u) {
      int r = ka + u;
      float yv = (u == 0 ? t0 : t1) + dsk * xf[u];
      float zv = zx[rb + OFF_Z + (size_t)h * 64 + r * 16 + p];
      float g  = zv / (1.f + __expf(-zv));                  // silu
      ybar[((size_t)b * SEQLEN + step) * D_INNER + h * 64 + r * 16 + p] = f2bf(yv * g);
    }
    __syncthreads();                                        // protect sBC for next step
  }
}

// ---------------- rmsnorm(hs + m_out) ----------------
__global__ __launch_bounds__(256)
void rmsnorm_kernel(const float* __restrict__ hs, const float* __restrict__ mo,
                    float* __restrict__ out) {
  __shared__ float red[256];
  const int row = blockIdx.x;
  const int tid = threadIdx.x;
  const size_t base = (size_t)row * D_MODEL;
  float v[4];
  float ss = 0.f;
#pragma unroll
  for (int i = 0; i < 4; ++i) {
    int c = tid + i * 256;
    v[i] = hs[base + c] + mo[base + c];
    ss += v[i] * v[i];
  }
  red[tid] = ss;
  __syncthreads();
  for (int s = 128; s > 0; s >>= 1) {
    if (tid < s) red[tid] += red[tid + s];
    __syncthreads();
  }
  float scale = rsqrtf(red[0] * (1.f / (float)D_MODEL) + 1e-5f);
#pragma unroll
  for (int i = 0; i < 4; ++i) out[base + tid + i * 256] = v[i] * scale;
}

// ---------------- launcher ----------------
extern "C" void kernel_launch(void* const* d_in, const int* in_sizes, int n_in,
                              void* d_out, int out_size, void* d_ws, size_t ws_size,
                              hipStream_t stream) {
  (void)in_sizes; (void)n_in; (void)out_size; (void)ws_size;
  const float* hidden = (const float*)d_in[0];
  const float* inj    = (const float*)d_in[1];
  const float* W_in   = (const float*)d_in[2];
  const float* W_out  = (const float*)d_in[3];
  const float* A_log  = (const float*)d_in[4];
  const float* dtb    = (const float*)d_in[5];
  const float* Dsk    = (const float*)d_in[6];
  float* out = (float*)d_out;

  char* ws = (char*)d_ws;
  size_t off = 0;
  auto take = [&](size_t bytes) -> char* {
    char* p = ws + off;
    off += (bytes + 255) & ~(size_t)255;
    return p;
  };
  float*          hs      = (float*)         take((size_t)NROWS * D_MODEL * 4);
  unsigned short* hs_bf   = (unsigned short*)take((size_t)NROWS * D_MODEL * 2);
  unsigned short* win_bf  = (unsigned short*)take((size_t)D_MODEL * NCOLS * 2);
  unsigned short* wout_bf = (unsigned short*)take((size_t)D_INNER * D_MODEL * 2);
  float*          zx      = (float*)         take((size_t)NROWS * NCOLS * 4);
  unsigned short* ybar    = (unsigned short*)take((size_t)NROWS * D_INNER * 2);
  float*          mo      = (float*)         take((size_t)NROWS * D_MODEL * 4);

  {
    int n = NROWS * D_MODEL;
    add_cvt_kernel<<<dim3((n + 255) / 256), dim3(256), 0, stream>>>(hidden, inj, hs, hs_bf, n);
  }
  {
    int n = D_MODEL * NCOLS;
    cvt_kernel<<<dim3((n + 255) / 256), dim3(256), 0, stream>>>(W_in, win_bf, n);
  }
  {
    int n = D_INNER * D_MODEL;
    cvt_kernel<<<dim3((n + 255) / 256), dim3(256), 0, stream>>>(W_out, wout_bf, n);
  }
  // zxbcdt = hs @ W_in : [8192,1024] x [1024,5152]
  gemm_bf16_wmma<<<dim3((NCOLS + 127) / 128, NROWS / 128), dim3(256), 0, stream>>>(
      hs_bf, win_bf, zx, NROWS, NCOLS, D_MODEL);
  // selective scan + fused gating
  scan_kernel<<<dim3(BATCH), dim3(1024), 0, stream>>>(zx, A_log, dtb, Dsk, ybar);
  // m_out = ybar @ W_out : [8192,2048] x [2048,1024]
  gemm_bf16_wmma<<<dim3(D_MODEL / 128, NROWS / 128), dim3(256), 0, stream>>>(
      ybar, wout_bf, mo, NROWS, D_MODEL, D_INNER);
  // out = rmsnorm(hs + m_out)
  rmsnorm_kernel<<<dim3(NROWS), dim3(256), 0, stream>>>(hs, mo, out);
}
